// fully_fused_mlp_56367150792832
// MI455X (gfx1250) — compile-verified, matrix operand order checked by reading, not run on hardware
//
#include <hip/hip_runtime.h>
#include <hip/hip_bf16.h>

typedef _Float16     h8_t  __attribute__((ext_vector_type(8)));
typedef _Float16     v16h  __attribute__((ext_vector_type(16)));
typedef _Float16     v8h   __attribute__((ext_vector_type(8)));
typedef float        v8f   __attribute__((ext_vector_type(8)));
typedef unsigned int u32x4 __attribute__((ext_vector_type(4)));

#define HASH_MASK  ((1u << 17) - 1u)
#define PRIME_Y    2654435761u

#define CHUNK   128                 // points per barrier phase
#define CHUNKS  4                   // chunks per block (weight staging amortized 4x)
#define PTS_PER_BLOCK (CHUNK * CHUNKS)

// Load one 16-half WMMA fragment (A or B operand) from an LDS row.
// Per CDNA5 ISA 16-bit A layout: lanes 0-15 take K {0..7,16..23}, lanes 16-31
// take K {8..15,24..31} of the 32-wide K block starting at `row`.
// Two contiguous 16B ds_load_b128 per lane.
__device__ __forceinline__ v16h load_frag(const _Float16* __restrict__ row, int hi) {
    h8_t lo = *(const h8_t*)(row + hi * 8);
    h8_t hh = *(const h8_t*)(row + 16 + hi * 8);
    v16h r;
#pragma unroll
    for (int i = 0; i < 8; ++i) { r[i] = lo[i]; r[8 + i] = hh[i]; }
    return r;
}

__device__ __forceinline__ v8f wmma_f32c(v16h a, v16h b, v8f c) {
    return __builtin_amdgcn_wmma_f32_16x16x32_f16(false, a, false, b,
                                                  (short)0, c, false, false);
}

// Single-instruction ReLU (avoid llvm.maxnum's canonicalize double-max).
__device__ __forceinline__ float relu1(float v) {
    float t;
    asm("v_max_num_f32 %0, %1, 0" : "=v"(t) : "v"(v));
    return t;
}

// Pack a float bias into a dword holding the f16 value in both halves.
__device__ __forceinline__ unsigned pack_bias_h2(float bias) {
    union { unsigned u; _Float16 h[2]; } cv;
    cv.h[0] = (_Float16)bias; cv.h[1] = cv.h[0];
    return cv.u;
}

#if __has_builtin(__builtin_amdgcn_wmma_f16_16x16x32_f16)
#define HAVE_WMMA_F16C 1
// f16 C/D variant: 4 accumulator VGPRs, no f32->f16 converts in the epilogue.
__device__ __forceinline__ v8h wmma_f16c(v16h a, v16h b, v8h c) {
    return __builtin_amdgcn_wmma_f16_16x16x32_f16(false, a, false, b,
                                                  (short)0, c, false, false);
}
// Packed ReLU on a row-pair dword: one v_pk_max_num_f16.
__device__ __forceinline__ unsigned pk_relu(unsigned v) {
    unsigned t;
    asm("v_pk_max_num_f16 %0, %1, 0" : "=v"(t) : "v"(v));
    return t;
}
#else
#define HAVE_WMMA_F16C 0
#endif

// Strides (in halfs) padded to stay 16B aligned while breaking bank alignment.
#define S32 40   // rows with 32 valid halfs
#define S64 72   // rows with 64 valid halfs

__global__ __launch_bounds__(256) void ngp_fused_mlp(
    const float* __restrict__ x,   const float* __restrict__ enc,
    const float* __restrict__ w0,  const float* __restrict__ b0,
    const float* __restrict__ w1,  const float* __restrict__ b1,
    const float* __restrict__ w2,  const float* __restrict__ b2,
    float* __restrict__ out, int N)
{
    __shared__ _Float16 act0[CHUNK * S32];   // hash features  [128][32]
    __shared__ _Float16 act1[CHUNK * S64];   // layer0 out     [128][64]
    __shared__ _Float16 act2[CHUNK * S64];   // layer1 out     [128][64]
    __shared__ _Float16 w0T [64 * S32];      // w0^T  [n=64][k=32]
    __shared__ _Float16 w1T [64 * S64];      // w1^T  [n=64][k=64]
    __shared__ _Float16 w2T [16 * S64];      // w2^T padded [n=16][k=64]
#if HAVE_WMMA_F16C
    __shared__ unsigned b0p[64], b1p[64];    // f16x2-packed biases (both halves)
#else
    __shared__ float b0p[64], b1p[64];
#endif
    __shared__ float b2s[16];

    const int tid = threadIdx.x;

    // ---- stage weights to LDS once per block (transposed, f16) ----
    for (int i = tid; i < 32 * 64; i += 256) {          // w0[k][n]
        int k = i >> 6, n = i & 63;
        w0T[n * S32 + k] = (_Float16)w0[i];
    }
    for (int i = tid; i < 64 * 64; i += 256) {          // w1[k][n]
        int k = i >> 6, n = i & 63;
        w1T[n * S64 + k] = (_Float16)w1[i];
    }
    for (int i = tid; i < 64 * 16; i += 256) {          // w2[k][3] zero-padded to 16 cols
        int k = i >> 4, n = i & 15;
        w2T[n * S64 + k] = (_Float16)((n < 3) ? w2[k * 3 + n] : 0.0f);
    }
    for (int i = tid; i < 64; i += 256) {
#if HAVE_WMMA_F16C
        b0p[i] = pack_bias_h2(b0[i]);
        b1p[i] = pack_bias_h2(b1[i]);
#else
        b0p[i] = b0[i];
        b1p[i] = b1[i];
#endif
    }
    if (tid < 16) b2s[tid] = (tid < 3) ? b2[tid] : 0.0f;

    const int wave  = tid >> 5;
    const int lane  = tid & 31;
    const int hi    = lane >> 4;     // K-half selector of the fragment layout
    const int ln    = lane & 15;     // row (A) / column (B,C,D) within tile
    const int rbase = wave * 16;

    for (int ch = 0; ch < CHUNKS; ++ch) {
        const int block0 = blockIdx.x * PTS_PER_BLOCK + ch * CHUNK;

        __syncthreads();   // act* from previous chunk fully consumed

        // ---- hash-grid encode: 2 threads per point, 8 levels each ----
        {
            const int p     = tid >> 1;        // 0..127 local point
            const int lhalf = tid & 1;         // level group 0: L0-7, 1: L8-15
            const int pg    = block0 + p;
            float px = 0.0f, py = 0.0f;
            if (pg < N) { px = x[pg * 2 + 0]; py = x[pg * 2 + 1]; }
            // 16 * 1.5^8 = 410.0625 exactly (3^k fits fp32 mantissa for k<=15)
            float s = lhalf ? 410.0625f : 16.0f;
#pragma unroll
            for (int l = 0; l < 8; ++l) {
                const int lev = lhalf * 8 + l;
                const float fx = px * s, fy = py * s;
                const float flx = floorf(fx), fly = floorf(fy);
                const float frx = fx - flx,  fry = fy - fly;
                const unsigned ix = (unsigned)flx, iy = (unsigned)fly;
                float e0 = 0.0f, e1 = 0.0f;
#pragma unroll
                for (int c = 0; c < 4; ++c) {
                    const unsigned cx = c & 1, cy = c >> 1;
                    const unsigned h   = (ix + cx) ^ ((iy + cy) * PRIME_Y);
                    const unsigned idx = h & HASH_MASK;
                    const float wx = cx ? frx : 1.0f - frx;
                    const float wy = cy ? fry : 1.0f - fry;
                    const float w  = wx * wy;
                    const float* e = enc + ((size_t)idx * 32u + (unsigned)lev * 2u);
                    e0 += w * e[0];           // global_load_b64 gather, L2 resident
                    e1 += w * e[1];
                }
                act0[p * S32 + lev * 2 + 0] = (_Float16)e0;
                act0[p * S32 + lev * 2 + 1] = (_Float16)e1;
                s *= 1.5f;
            }
        }
        __syncthreads();   // encode (own wave's rows) + weights visible; waves private now

        // ---- fused MLP on WMMA, one wave = one 16-row M tile ----

        // Layer 0: [16x32] @ [32x64] + b0, ReLU -> act1
        {
            const v16h a = load_frag(&act0[(rbase + ln) * S32], hi);
#pragma unroll
            for (int nt = 0; nt < 4; ++nt) {
                const v16h b = load_frag(&w0T[(nt * 16 + ln) * S32], hi);
#if HAVE_WMMA_F16C
                const unsigned bb = b0p[nt * 16 + ln];
                u32x4 uu = { bb, bb, bb, bb };
                v8h c = __builtin_bit_cast(v8h, uu);
                c = wmma_f16c(a, b, c);
                u32x4 u = __builtin_bit_cast(u32x4, c);
#pragma unroll
                for (int j = 0; j < 4; ++j) {
                    union { unsigned u; _Float16 h[2]; } cv;
                    cv.u = pk_relu(u[j]);              // rows 2j, 2j+1 (+8*hi)
                    act1[(rbase + 2 * j     + 8 * hi) * S64 + nt * 16 + ln] = cv.h[0];
                    act1[(rbase + 2 * j + 1 + 8 * hi) * S64 + nt * 16 + ln] = cv.h[1];
                }
#else
                const float bias = b0p[nt * 16 + ln];
                v8f c;
#pragma unroll
                for (int r = 0; r < 8; ++r) c[r] = bias;
                c = wmma_f32c(a, b, c);
#pragma unroll
                for (int r = 0; r < 8; ++r)
                    act1[(rbase + r + 8 * hi) * S64 + nt * 16 + ln] =
                        (_Float16)relu1(c[r]);
#endif
            }
        }

        // Layer 1: [16x64] @ [64x64] + b1, ReLU -> act2 (two chained K=32 WMMAs)
        {
            const v16h alo = load_frag(&act1[(rbase + ln) * S64],      hi);
            const v16h ahi = load_frag(&act1[(rbase + ln) * S64 + 32], hi);
#pragma unroll
            for (int nt = 0; nt < 4; ++nt) {
                const v16h blo = load_frag(&w1T[(nt * 16 + ln) * S64],      hi);
                const v16h bhi = load_frag(&w1T[(nt * 16 + ln) * S64 + 32], hi);
#if HAVE_WMMA_F16C
                const unsigned bb = b1p[nt * 16 + ln];
                u32x4 uu = { bb, bb, bb, bb };
                v8h c = __builtin_bit_cast(v8h, uu);
                c = wmma_f16c(alo, blo, c);
                c = wmma_f16c(ahi, bhi, c);
                u32x4 u = __builtin_bit_cast(u32x4, c);
#pragma unroll
                for (int j = 0; j < 4; ++j) {
                    union { unsigned u; _Float16 h[2]; } cv;
                    cv.u = pk_relu(u[j]);
                    act2[(rbase + 2 * j     + 8 * hi) * S64 + nt * 16 + ln] = cv.h[0];
                    act2[(rbase + 2 * j + 1 + 8 * hi) * S64 + nt * 16 + ln] = cv.h[1];
                }
#else
                const float bias = b1p[nt * 16 + ln];
                v8f c;
#pragma unroll
                for (int r = 0; r < 8; ++r) c[r] = bias;
                c = wmma_f32c(alo, blo, c);
                c = wmma_f32c(ahi, bhi, c);
#pragma unroll
                for (int r = 0; r < 8; ++r)
                    act2[(rbase + r + 8 * hi) * S64 + nt * 16 + ln] =
                        (_Float16)relu1(c[r]);
#endif
            }
        }

        // Layer 2: [16x64] @ [64x16(pad of 3)] + b2 -> global out (f32 accumulate)
        {
            const v16h alo = load_frag(&act2[(rbase + ln) * S64],      hi);
            const v16h ahi = load_frag(&act2[(rbase + ln) * S64 + 32], hi);
            const v16h blo = load_frag(&w2T[ln * S64],      hi);
            const v16h bhi = load_frag(&w2T[ln * S64 + 32], hi);
            v8f c;
            const float bias = b2s[ln];
#pragma unroll
            for (int r = 0; r < 8; ++r) c[r] = bias;
            c = wmma_f32c(alo, blo, c);
            c = wmma_f32c(ahi, bhi, c);
            if (ln < 3) {
#pragma unroll
                for (int r = 0; r < 8; ++r) {
                    const int row = block0 + rbase + r + 8 * hi;
                    if (row < N) out[row * 3 + ln] = c[r];
                }
            }
        }
    }
}

extern "C" void kernel_launch(void* const* d_in, const int* in_sizes, int n_in,
                              void* d_out, int out_size, void* d_ws, size_t ws_size,
                              hipStream_t stream) {
    const float* x   = (const float*)d_in[0];
    const float* enc = (const float*)d_in[1];
    const float* w0  = (const float*)d_in[2];
    const float* b0  = (const float*)d_in[3];
    const float* w1  = (const float*)d_in[4];
    const float* b1  = (const float*)d_in[5];
    const float* w2  = (const float*)d_in[6];
    const float* b2  = (const float*)d_in[7];
    float* out = (float*)d_out;

    const int N = in_sizes[0] / 2;                       // x is [N,2]
    const int blocks = (N + PTS_PER_BLOCK - 1) / PTS_PER_BLOCK;
    ngp_fused_mlp<<<blocks, 256, 0, stream>>>(x, enc, w0, b0, w1, b1, w2, b2, out, N);
}
